// VisualOrientationPosterior_38843684225527
// MI455X (gfx1250) — matrix operand order repaired and checked by reading, not compiled, not run on hardware
//
#include <hip/hip_runtime.h>
#include <cstdint>
#include <cstddef>

typedef __attribute__((ext_vector_type(16))) _Float16 v16h;
typedef __attribute__((ext_vector_type(8)))  _Float16 v8h;
typedef __attribute__((ext_vector_type(8)))  float    v8f;
typedef __attribute__((ext_vector_type(4)))  int      v4i;

#define WMMA16(A,Bv,C) __builtin_amdgcn_wmma_f32_16x16x32_f16(false,(A),false,(Bv),(short)0,(C),false,false)

// CDNA5 async global->LDS copies (ASYNCcnt-tracked), with portable fallback.
#if defined(__AMDGCN__) && __has_builtin(__builtin_amdgcn_global_load_async_to_lds_b128) && __has_builtin(__builtin_amdgcn_s_wait_asynccnt)
#define HAVE_ASYNC_LDS 1
typedef v4i __attribute__((address_space(1))) as1_v4i;   // global
typedef v4i __attribute__((address_space(3))) as3_v4i;   // LDS
#endif

__device__ __forceinline__ void lds_copy_b128(void* ldst, const void* gsrc) {
#ifdef HAVE_ASYNC_LDS
    __builtin_amdgcn_global_load_async_to_lds_b128(
        (as1_v4i*)gsrc, (as3_v4i*)ldst, 0, 0);
#else
    *(uint4*)ldst = *(const uint4*)gsrc;
#endif
}
__device__ __forceinline__ void lds_copy_wait() {
#ifdef HAVE_ASYNC_LDS
    __builtin_amdgcn_s_wait_asynccnt(0);
#endif
}

// Problem constants
static constexpr int NB   = 8;     // batch
static constexpr int CIN  = 768;
static constexpr int HID  = 128;
static constexpr int NA   = 36;    // angles
static constexpr int HW   = 32;
static constexpr int NPIX = 1024;  // 32*32

// ---------------------------------------------------------------------------
// Pack a row-major f32 weight [M=MT*16][srcK] into WMMA A-operand tiles.
//   M = mt*16 + (lane&15)
//   K(h) = kt*32 + 16*(lane>=16) + h
// dst tile index = mt*KT + kt, tile stride = 512 halves.
// ---------------------------------------------------------------------------
__global__ __launch_bounds__(32) void pack_a_kernel(const float* __restrict__ src,
                                                    int srcK, int colOff, int KT,
                                                    _Float16* __restrict__ dst) {
    int tile = blockIdx.x;
    int mt = tile / KT, kt = tile % KT;
    int lane = threadIdx.x;
    int hl = lane >> 4;
    int M = mt * 16 + (lane & 15);
    _Float16 o[16];
#pragma unroll
    for (int h = 0; h < 16; ++h) {
        int K = kt * 32 + hl * 16 + h;
        o[h] = (_Float16)src[(size_t)M * srcK + colOff + K];
    }
    *(v16h*)(dst + ((size_t)tile * 32 + lane) * 16) = *(v16h*)o;
}

// ---------------------------------------------------------------------------
// Bilinear rotation taps per (angle, pixel): 4 gather indices + weights
// (affine_grid + grid_sample, align_corners=False, zeros padding).
// ---------------------------------------------------------------------------
__global__ __launch_bounds__(256) void rot_build_kernel(const float* __restrict__ angles,
                                                        int4*  __restrict__ rotIdx,
                                                        float4* __restrict__ rotW) {
    int e = blockIdx.x * blockDim.x + threadIdx.x;
    if (e >= NA * NPIX) return;
    int a = e >> 10, p = e & 1023;
    int py = p >> 5, px = p & 31;
    float th = angles[a] * 0.017453292519943295f;
    float c = cosf(th), s = sinf(th);
    float xn = (2.0f * px + 1.0f) / (float)HW - 1.0f;
    float yn = (2.0f * py + 1.0f) / (float)HW - 1.0f;
    float gx = c * xn - s * yn;
    float gy = s * xn + c * yn;
    float ix = ((gx + 1.0f) * HW - 1.0f) * 0.5f;
    float iy = ((gy + 1.0f) * HW - 1.0f) * 0.5f;
    float x0 = floorf(ix), y0 = floorf(iy);
    float wx = ix - x0, wy = iy - y0;
    int x0i = (int)x0, y0i = (int)y0;
    int xs[4] = { x0i, x0i + 1, x0i,     x0i + 1 };
    int ys[4] = { y0i, y0i,     y0i + 1, y0i + 1 };
    float wv[4] = { (1.f - wx) * (1.f - wy), wx * (1.f - wy),
                    (1.f - wx) * wy,         wx * wy };
    int   idx[4];
    float wgt[4];
#pragma unroll
    for (int k = 0; k < 4; ++k) {
        bool valid = (xs[k] >= 0) & (xs[k] < HW) & (ys[k] >= 0) & (ys[k] < HW);
        int xc = min(max(xs[k], 0), HW - 1);
        int yc = min(max(ys[k], 0), HW - 1);
        idx[k] = yc * HW + xc;
        wgt[k] = valid ? wv[k] : 0.0f;
    }
    rotIdx[e] = make_int4(idx[0], idx[1], idx[2], idx[3]);
    rotW[e]   = make_float4(wgt[0], wgt[1], wgt[2], wgt[3]);
}

// ---------------------------------------------------------------------------
// Projection + per-pixel L2 norm:  P = Wx @ map  ->  l2norm over 128 channels
// grid (64 pixel-tiles, NB, 2 maps), block 128 (4 waves).
// Output f16 [b][pixel][128] (pixel-major for rotation gathers).
// ---------------------------------------------------------------------------
__global__ __launch_bounds__(128) void proj_kernel(const float* __restrict__ gmap,
                                                   const float* __restrict__ qmap,
                                                   const _Float16* __restrict__ WgP,
                                                   const _Float16* __restrict__ WqP,
                                                   _Float16* __restrict__ g16,
                                                   _Float16* __restrict__ q16) {
    __shared__ _Float16 inT[16 * CIN];   // 24 KB, pixel-major f16 slab
    __shared__ float    sq[4 * 16];      // per-(wave, pixel) sum of squares

    int b  = blockIdx.y;
    int p0 = blockIdx.x * 16;
    int which = blockIdx.z;
    const float*    src = which ? qmap : gmap;
    const _Float16* AP  = which ? WqP  : WgP;
    _Float16*       dst = which ? q16  : g16;

    int t = threadIdx.x;
    {   // load input tile [CIN x 16 px] -> LDS f16 [px][CIN]
        int px = t & 15, c0 = t >> 4;  // c0 in 0..7
        const float* sp = src + (size_t)b * CIN * NPIX + p0 + px;
        for (int c = c0; c < CIN; c += 8)
            inT[px * CIN + c] = (_Float16)sp[(size_t)c * NPIX];
    }
    __syncthreads();

    int lane = t & 31, w = t >> 5, hl = lane >> 4, pix = lane & 15;
    v8f cc[2];
#pragma unroll
    for (int i = 0; i < 2; ++i) {
        int mt = w + 4 * i;
        v8f c = {};
        for (int kt = 0; kt < 24; ++kt) {
            v16h av = *(const v16h*)(AP + ((size_t)(mt * 24 + kt) * 32 + lane) * 16);
            v16h bv = *(const v16h*)&inT[pix * CIN + kt * 32 + hl * 16];
            c = WMMA16(av, bv, c);
        }
        cc[i] = c;
    }
    // deterministic per-pixel sum of squares over all 128 channels
    float s = 0.f;
#pragma unroll
    for (int i = 0; i < 2; ++i)
#pragma unroll
        for (int v = 0; v < 8; ++v) s += cc[i][v] * cc[i][v];
    s += __shfl_xor(s, 16, 32);          // combine lane halves (hl 0/1)
    if (lane < 16) sq[w * 16 + pix] = s; // per-wave partial
    __syncthreads();
    float ns = sq[pix] + sq[16 + pix] + sq[32 + pix] + sq[48 + pix];
    float scale = 1.0f / fmaxf(sqrtf(ns), 1e-12f);
#pragma unroll
    for (int i = 0; i < 2; ++i) {
        int mt = w + 4 * i;
        _Float16 o[8];
#pragma unroll
        for (int v = 0; v < 8; ++v) o[v] = (_Float16)(cc[i][v] * scale);
        *(v8h*)(dst + ((size_t)(b * NPIX + p0 + pix)) * HID + mt * 16 + hl * 8) = *(v8h*)o;
    }
}

// ---------------------------------------------------------------------------
// baseb = W1g @ g + b1, stored in C-tile layout:
//   baseb[ ((b*8 + mt)*64 + nt)*256 + lane*8 + v ]   (lane-major, v contiguous)
// so the posterior kernel's accumulator init is one 32B vector load per lane.
// ---------------------------------------------------------------------------
__global__ __launch_bounds__(128) void base_kernel(const _Float16* __restrict__ g16,
                                                   const _Float16* __restrict__ W1gP,
                                                   const float* __restrict__ b1,
                                                   float* __restrict__ baseb) {
    int b  = blockIdx.y;
    int nt = blockIdx.x;
    int p0 = nt * 16;
    int t = threadIdx.x;
    int lane = t & 31, w = t >> 5, hl = lane >> 4, pix = lane & 15;
#pragma unroll
    for (int i = 0; i < 2; ++i) {
        int mt = w + 4 * i;
        v8f c = {};
#pragma unroll
        for (int kt = 0; kt < 4; ++kt) {
            v16h av = *(const v16h*)(W1gP + ((size_t)(mt * 4 + kt) * 32 + lane) * 16);
            v16h bv = *(const v16h*)(g16 + ((size_t)(b * NPIX + p0 + pix)) * HID
                                          + kt * 32 + hl * 16);
            c = WMMA16(av, bv, c);
        }
        v8f o;
#pragma unroll
        for (int v = 0; v < 8; ++v) o[v] = c[v] + b1[mt * 16 + hl * 8 + v];
        *(v8f*)(baseb + (((size_t)(b * 8 + mt) * 64 + nt) * 256 + lane * 8)) = o;
    }
}

// ---------------------------------------------------------------------------
// Main stage: one WG per (angle a, batch b).  h = relu(baseb + W1q*qr +
// W1m*(g*qr) + W1a*|g-qr|);  out[b][a] = mean_pix( W2 . h ) + b2.
// Dynamic LDS: W1cat packed (96 KB, async-loaded) + g/qr tiles + reduction.
// ---------------------------------------------------------------------------
__global__ __launch_bounds__(128) void posterior_kernel(const _Float16* __restrict__ g16,
                                                        const _Float16* __restrict__ q16,
                                                        const _Float16* __restrict__ W1cP,
                                                        const float* __restrict__ baseb,
                                                        const float* __restrict__ W2,
                                                        const float* __restrict__ b2,
                                                        const int4*  __restrict__ rotIdx,
                                                        const float4* __restrict__ rotW,
                                                        float* __restrict__ out) {
    extern __shared__ unsigned char smem[];
    _Float16* w1s = (_Float16*)smem;          // 8*12*512 = 49152 halves (96 KB)
    _Float16* gT  = w1s + 49152;              // 16 px * 128 ch
    _Float16* qT  = gT + 2048;                // 16 px * 128 ch
    float*    red = (float*)(qT + 2048);      // 128

    int a = blockIdx.x, b = blockIdx.y;
    int t = threadIdx.x;
    int lane = t & 31, w = t >> 5, hl = lane >> 4, pix = lane & 15;
    int gpix = t & 15, grp = t >> 4;          // staging role: 16 px x 8 ch-groups

    {   // preload packed W1cat into LDS (6144 x b128, ASYNCcnt-tracked)
        const uint4* s = (const uint4*)W1cP;
        uint4* d = (uint4*)w1s;
        for (int i = t; i < 6144; i += 128) lds_copy_b128(&d[i], &s[i]);
        lds_copy_wait();
    }

    // W2 coefficients for this wave's two M-tiles, hoisted into registers
    v8f w2r[2];
#pragma unroll
    for (int i = 0; i < 2; ++i)
        w2r[i] = *(const v8f*)(W2 + (w + 4 * i) * 16 + hl * 8);

    float acc = 0.0f;

    for (int nt = 0; nt < 64; ++nt) {
        int p0 = nt * 16;
        __syncthreads();   // previous-iteration LDS reads done (also covers preload)
        {   // stage g tile (async copy) and rotated-q tile into LDS [px][128]
            const uint4* gs = (const uint4*)(g16 + ((size_t)(b * NPIX + p0 + gpix)) * HID
                                                  + grp * 16);
            uint4* gd = (uint4*)&gT[gpix * HID + grp * 16];
            lds_copy_b128(&gd[0], &gs[0]);
            lds_copy_b128(&gd[1], &gs[1]);
            if (nt < 63)   // warm next tile (global_prefetch_b8)
                __builtin_prefetch(g16 + ((size_t)(b * NPIX + p0 + 16 + gpix)) * HID + grp * 16, 0, 0);

            int e = a * NPIX + p0 + gpix;
            int4   id = rotIdx[e];
            float4 ww = rotW[e];
            int   ids[4] = { id.x, id.y, id.z, id.w };
            float wv[4]  = { ww.x, ww.y, ww.z, ww.w };
            float aq[16];
#pragma unroll
            for (int j = 0; j < 16; ++j) aq[j] = 0.0f;
#pragma unroll
            for (int k = 0; k < 4; ++k) {
                const _Float16* qs = q16 + ((size_t)(b * NPIX + ids[k])) * HID + grp * 16;
#pragma unroll
                for (int j = 0; j < 16; ++j) aq[j] += wv[k] * (float)qs[j];
            }
            _Float16 qh[16];
#pragma unroll
            for (int j = 0; j < 16; ++j) qh[j] = (_Float16)aq[j];
            uint4* qd = (uint4*)&qT[gpix * HID + grp * 16];
            qd[0] = ((uint4*)qh)[0]; qd[1] = ((uint4*)qh)[1];
        }
        lds_copy_wait();   // g-tile async copies landed before the barrier
        __syncthreads();

#pragma unroll
        for (int i = 0; i < 2; ++i) {
            int mt = w + 4 * i;
            // C init = baseb tile (lane-major layout -> one 32B vector load)
            v8f c = *(const v8f*)(baseb + (((size_t)(b * 8 + mt) * 64 + nt) * 256
                                           + lane * 8));
            for (int kt = 0; kt < 12; ++kt) {
                int coff = (kt & 3) * 32 + hl * 16;
                v16h qv = *(const v16h*)&qT[pix * HID + coff];
                v16h xv;
                if (kt < 4) {
                    xv = qv;                                   // W1q . qr
                } else {
                    v16h gv = *(const v16h*)&gT[pix * HID + coff];
                    if (kt < 8) {
#pragma unroll
                        for (int j = 0; j < 16; ++j) xv[j] = gv[j] * qv[j];   // g*qr
                    } else {
#pragma unroll
                        for (int j = 0; j < 16; ++j) {
                            _Float16 d = gv[j] - qv[j];
                            xv[j] = (d < (_Float16)0.0f) ? (_Float16)(-d) : d; // |g-qr|
                        }
                    }
                }
                v16h av = *(const v16h*)&w1s[((size_t)(mt * 12 + kt) * 32 + lane) * 16];
                c = WMMA16(av, xv, c);
            }
            // relu + W2 dot, accumulate scalar partial
#pragma unroll
            for (int v = 0; v < 8; ++v)
                acc += fmaxf(c[v], 0.0f) * w2r[i][v];
        }
    }

    red[t] = acc;
    __syncthreads();
    for (int s2 = 64; s2 > 0; s2 >>= 1) {
        if (t < s2) red[t] += red[t + s2];
        __syncthreads();
    }
    if (t == 0) out[b * NA + a] = red[0] * (1.0f / (float)NPIX) + b2[0];
}

// ---------------------------------------------------------------------------
extern "C" void kernel_launch(void* const* d_in, const int* in_sizes, int n_in,
                              void* d_out, int out_size, void* d_ws, size_t ws_size,
                              hipStream_t stream) {
    (void)in_sizes; (void)n_in; (void)out_size; (void)ws_size;
    const float* gallery = (const float*)d_in[0];
    const float* query   = (const float*)d_in[1];
    const float* Wg      = (const float*)d_in[2];
    const float* Wq      = (const float*)d_in[3];
    const float* W1      = (const float*)d_in[4];
    const float* b1      = (const float*)d_in[5];
    const float* W2      = (const float*)d_in[6];
    const float* b2      = (const float*)d_in[7];
    const float* angles  = (const float*)d_in[8];
    float* out = (float*)d_out;

    // workspace carve-up (256 B aligned)
    uint8_t* ws = (uint8_t*)d_ws;
    size_t off = 0;
    auto carve = [&](size_t bytes) -> uint8_t* {
        uint8_t* p = ws + off;
        off += (bytes + 255) & ~(size_t)255;
        return p;
    };
    _Float16* WgP  = (_Float16*)carve(8 * 24 * 512 * sizeof(_Float16)); // 192 KB
    _Float16* WqP  = (_Float16*)carve(8 * 24 * 512 * sizeof(_Float16));
    _Float16* W1gP = (_Float16*)carve(8 *  4 * 512 * sizeof(_Float16));
    _Float16* W1cP = (_Float16*)carve(8 * 12 * 512 * sizeof(_Float16));
    _Float16* g16  = (_Float16*)carve((size_t)NB * NPIX * HID * sizeof(_Float16)); // 2 MB
    _Float16* q16  = (_Float16*)carve((size_t)NB * NPIX * HID * sizeof(_Float16));
    float*    baseb= (float*)   carve((size_t)NB * HID * NPIX * sizeof(float));    // 4 MB
    int4*     rotI = (int4*)    carve((size_t)NA * NPIX * sizeof(int4));
    float4*   rotWt= (float4*)  carve((size_t)NA * NPIX * sizeof(float4));

    // 1. pack weights into WMMA A-operand tile layout
    pack_a_kernel<<<8 * 24, 32, 0, stream>>>(Wg, CIN, 0,   24, WgP);
    pack_a_kernel<<<8 * 24, 32, 0, stream>>>(Wq, CIN, 0,   24, WqP);
    pack_a_kernel<<<8 *  4, 32, 0, stream>>>(W1, 512, 0,    4, W1gP);   // W1g
    pack_a_kernel<<<8 * 12, 32, 0, stream>>>(W1, 512, 128, 12, W1cP);   // [W1q|W1m|W1a]

    // 2. rotation tap table
    rot_build_kernel<<<(NA * NPIX + 255) / 256, 256, 0, stream>>>(angles, rotI, rotWt);

    // 3. projections + L2 norm (z=0 gallery->g16, z=1 query->q16)
    proj_kernel<<<dim3(64, NB, 2), 128, 0, stream>>>(gallery, query, WgP, WqP, g16, q16);

    // 4. baseb = W1g@g + b1 (C-tile layout)
    base_kernel<<<dim3(64, NB), 128, 0, stream>>>(g16, W1gP, b1, baseb);

    // 5. main posterior: one WG per (angle, batch)
    size_t lds = (49152 + 2048 + 2048) * sizeof(_Float16) + 128 * sizeof(float);
    posterior_kernel<<<dim3(NA, NB), 128, lds, stream>>>(g16, q16, W1cP, baseb,
                                                         W2, b2, rotI, rotWt, out);
}
